// Transformer_53291954209073
// MI455X (gfx1250) — matrix operand order repaired and checked by reading, not compiled, not run on hardware
//
#include <hip/hip_runtime.h>
#include <hip/hip_bf16.h>

// ---------------------------------------------------------------------------
// Types for WMMA fragments (CDNA5 / gfx1250, wave32)
// ---------------------------------------------------------------------------
typedef __attribute__((ext_vector_type(16))) _Float16 v16h;
typedef __attribute__((ext_vector_type(8)))  _Float16 v8h;
typedef __attribute__((ext_vector_type(8)))  float    v8f;

#define DEPTH 2
#define DIM   128
#define VOCAB 16384
#define BATCH 2
#define SEQ   2048
#define MROWS (BATCH * SEQ)   // 4096
#define FFD   512
#define CHK   64              // scan chunk length
#define NCH   (SEQ / CHK)     // 32 chunks per batch

enum { EPI_BIAS = 1, EPI_GELU = 2, EPI_RES = 4, EPI_F16 = 8, EPI_F32 = 16 };

// ---------------------------------------------------------------------------
// Embedding gather: x[row] = emb[token[row]]
// ---------------------------------------------------------------------------
__global__ void k_embed(const int* __restrict__ tokens,
                        const float* __restrict__ emb,
                        float* __restrict__ x) {
  int i = blockIdx.x * 256 + threadIdx.x;            // over 4096*32 float4s
  if (i >= MROWS * (DIM / 4)) return;
  int row = i >> 5, c4 = i & 31;
  int t = tokens[row];
  ((float4*)x)[(size_t)row * 32 + c4] = ((const float4*)emb)[(size_t)t * 32 + c4];
}

// ---------------------------------------------------------------------------
// RMSNorm per row (DIM=128) -> f16 output.  8 waves/block, one wave per row.
// ---------------------------------------------------------------------------
__global__ void k_rmsnorm(const float* __restrict__ x,
                          const float* __restrict__ gamma,
                          _Float16* __restrict__ h) {
  int row  = blockIdx.x * 8 + (threadIdx.x >> 5);
  int lane = threadIdx.x & 31;
  const float* xr = x + (size_t)row * DIM;
  float4 xv = *(const float4*)(xr + lane * 4);
  float ss = xv.x * xv.x + xv.y * xv.y + xv.z * xv.z + xv.w * xv.w;
  ss += __shfl_xor(ss, 1, 32);
  ss += __shfl_xor(ss, 2, 32);
  ss += __shfl_xor(ss, 4, 32);
  ss += __shfl_xor(ss, 8, 32);
  ss += __shfl_xor(ss, 16, 32);
  float sc = 11.313708498984761f / fmaxf(sqrtf(ss), 1e-12f);  // sqrt(128)/||x||
  float4 gv = *(const float4*)(gamma + lane * 4);
  _Float16* hr = h + (size_t)row * DIM + lane * 4;
  hr[0] = (_Float16)(xv.x * sc * gv.x);
  hr[1] = (_Float16)(xv.y * sc * gv.y);
  hr[2] = (_Float16)(xv.z * sc * gv.z);
  hr[3] = (_Float16)(xv.w * sc * gv.w);
}

// ---------------------------------------------------------------------------
// Pack f32 weight W[K][N] (row-major) into the per-lane WMMA B-fragment
// layout as f16 (lane = column, 16 consecutive K per lane; coalesced b128s).
// ---------------------------------------------------------------------------
__global__ void k_packB(const float* __restrict__ W, _Float16* __restrict__ out,
                        int K, int N) {
  int KB = K >> 5;
  int total = (N >> 4) * KB * 32 * 8;
  int idx = blockIdx.x * 256 + threadIdx.x;
  if (idx >= total) return;
  int dw   = idx & 7;
  int lane = (idx >> 3) & 31;
  int rest = idx >> 8;
  int kb = rest % KB;
  int np = rest / KB;
  int k  = kb * 32 + (lane >> 4) * 16 + dw * 2;
  int n  = np * 16 + (lane & 15);
  out[(size_t)idx * 2 + 0] = (_Float16)W[(size_t)k * N + n];
  out[(size_t)idx * 2 + 1] = (_Float16)W[(size_t)(k + 1) * N + n];
}

// ---------------------------------------------------------------------------
// WMMA GEMM: C[M][N] = A_f16[M][K] @ Bpacked + epilogue.
// Block = 128 threads = 4 waves; each wave computes NT 16x16 tiles (shared A
// fragment amortized over NT WMMAs).  grid = (N/(64*NT), M/16).
// ---------------------------------------------------------------------------
template <int NT>
__global__ void k_gemm(const _Float16* __restrict__ A,
                       const _Float16* __restrict__ Bp,
                       const float* __restrict__ bias,
                       const float* __restrict__ resid,
                       float* __restrict__ C32,
                       _Float16* __restrict__ C16,
                       int M, int N, int K, int epi) {
  const int lane = threadIdx.x & 31;
  const int wave = threadIdx.x >> 5;
  const int m0 = blockIdx.y * 16;
  const int nt0 = (blockIdx.x * 4 + wave) * NT;   // first 16-column panel
  const int KB = K >> 5;
  const int row = lane & 15, kh = lane >> 4;

  const _Float16* arow  = A + (size_t)(m0 + row) * K + kh * 16;
  const _Float16* bbase = Bp + (size_t)nt0 * KB * 512 + lane * 16;

  v8f acc[NT];
#pragma unroll
  for (int j = 0; j < NT; ++j) acc[j] = (v8f){};

  for (int kb = 0; kb < KB; ++kb) {
    v8h alo = *(const v8h*)(arow + kb * 32);
    v8h ahi = *(const v8h*)(arow + kb * 32 + 8);
    if (kb + 1 < KB)
      __builtin_prefetch(bbase + (size_t)(kb + 1) * 512, 0, 1);
    v16h a;
#pragma unroll
    for (int i = 0; i < 8; ++i) { a[i] = alo[i]; a[i + 8] = ahi[i]; }
#pragma unroll
    for (int j = 0; j < NT; ++j) {
      v16h b = *(const v16h*)(bbase + ((size_t)j * KB + kb) * 512);
      acc[j] = __builtin_amdgcn_wmma_f32_16x16x32_f16(
          /*neg_a=*/false, a, /*neg_b=*/false, b,
          /*c_mod=*/(short)0, acc[j], /*reuse_a=*/false, /*reuse_b=*/false);
    }
  }

  // C/D layout: lane -> column; VGPR v -> row m0 + v + 8*(lane>>4)
  const int mb = m0 + (lane >> 4) * 8;
#pragma unroll
  for (int j = 0; j < NT; ++j) {
    const int cn = (nt0 + j) * 16 + (lane & 15);
    float bv = (epi & EPI_BIAS) ? bias[cn] : 0.0f;
#pragma unroll
    for (int v = 0; v < 8; ++v) {
      int m = mb + v;
      float val = acc[j][v] + bv;
      if (epi & EPI_GELU) val = 0.5f * val * (1.0f + erff(val * 0.70710678118654752f));
      if (epi & EPI_RES)  val += resid[(size_t)m * N + cn];
      if (epi & EPI_F32)  C32[(size_t)m * N + cn] = val;
      if (epi & EPI_F16)  C16[(size_t)m * N + cn] = (_Float16)val;
    }
  }
}

// ---------------------------------------------------------------------------
// Gate decay:  g = Re(sigmoid(|a2|)*e^{i*angle(a2)}) = sigmoid(r)*ar/r
// (r==0 -> 0.5).  abuf is [4096][256] interleaved (re,im).
// ---------------------------------------------------------------------------
__global__ void k_gate(const float* __restrict__ abuf, float* __restrict__ g) {
  int i = blockIdx.x * 256 + threadIdx.x;            // over 4096*128
  if (i >= MROWS * DIM) return;
  float ar = abuf[(size_t)2 * i];
  float ai = abuf[(size_t)2 * i + 1];
  float r  = sqrtf(ar * ar + ai * ai);
  float s  = 1.0f / (1.0f + expf(-r));
  g[i] = (r > 0.0f) ? (s * ar / r) : 0.5f;
}

// ---------------------------------------------------------------------------
// Chunked gate-loop scan (exact w.r.t. the sequential recurrence):
//   S_t = g_t (*) S_{t-1} + k_t v_t^T ,  y_t = q_t . S_t[:,e]
// Pass 1: per (b,chunk,egroup) block, zero-init chunk-local scan.
//   Emits y_local, chunk final state S_c[d][e], decay product P_c[d].
// Block = 256 thr = 8 waves; wave = one e column; lane owns 4 d's.
// grid = BATCH*NCH*16.
// ---------------------------------------------------------------------------
__global__ void k_scan1(const float* __restrict__ qkv,   // [4096][384] q|k|v
                        const float* __restrict__ g,     // [4096][128]
                        float* __restrict__ ybuf,        // [4096][128]
                        float* __restrict__ Sc,          // [B][NCH][128][128]
                        float* __restrict__ Pc) {        // [B][NCH][128]
  const int eg = blockIdx.x & 15;
  const int c  = (blockIdx.x >> 4) & (NCH - 1);
  const int b  = blockIdx.x >> 9;
  const int wv = threadIdx.x >> 5;
  const int ln = threadIdx.x & 31;
  const int e  = eg * 8 + wv;
  const int d0 = ln * 4;
  float s0 = 0.f, s1 = 0.f, s2 = 0.f, s3 = 0.f;
  float p0 = 1.f, p1 = 1.f, p2 = 1.f, p3 = 1.f;
  const size_t rbase = (size_t)b * SEQ + (size_t)c * CHK;
  for (int t = 0; t < CHK; ++t) {
    const float* r  = qkv + (rbase + t) * 384;
    const float* gr = g   + (rbase + t) * 128;
    float4 qv = *(const float4*)(r + d0);
    float4 kv = *(const float4*)(r + 128 + d0);
    float4 gv = *(const float4*)(gr + d0);
    float  vv = r[256 + e];
    s0 = fmaf(gv.x, s0, kv.x * vv);
    s1 = fmaf(gv.y, s1, kv.y * vv);
    s2 = fmaf(gv.z, s2, kv.z * vv);
    s3 = fmaf(gv.w, s3, kv.w * vv);
    p0 *= gv.x; p1 *= gv.y; p2 *= gv.z; p3 *= gv.w;
    float acc = qv.x * s0 + qv.y * s1 + qv.z * s2 + qv.w * s3;
    acc += __shfl_xor(acc, 1, 32);
    acc += __shfl_xor(acc, 2, 32);
    acc += __shfl_xor(acc, 4, 32);
    acc += __shfl_xor(acc, 8, 32);
    acc += __shfl_xor(acc, 16, 32);
    if (ln == 0) ybuf[(rbase + t) * 128 + e] = acc;
  }
  const size_t sb = (size_t)(b * NCH + c) * 128;
  Sc[(sb + d0 + 0) * 128 + e] = s0;
  Sc[(sb + d0 + 1) * 128 + e] = s1;
  Sc[(sb + d0 + 2) * 128 + e] = s2;
  Sc[(sb + d0 + 3) * 128 + e] = s3;
  if (eg == 0 && wv == 0) {
    Pc[sb + d0 + 0] = p0;
    Pc[sb + d0 + 1] = p1;
    Pc[sb + d0 + 2] = p2;
    Pc[sb + d0 + 3] = p3;
  }
}

// ---------------------------------------------------------------------------
// Pass 2: sequential combine over the NCH chunks (per (b,d,e) thread):
//   Sin[c] = running;  running = P_c[d]*running + S_c[d][e]
// ---------------------------------------------------------------------------
__global__ void k_scan2(const float* __restrict__ Sc,
                        const float* __restrict__ Pc,
                        float* __restrict__ Sin) {
  int i = blockIdx.x * 256 + threadIdx.x;            // over B*128*128
  if (i >= BATCH * DIM * DIM) return;
  int e = i & 127;
  int d = (i >> 7) & 127;
  int b = i >> 14;
  float s = 0.f;
  for (int c = 0; c < NCH; ++c) {
    size_t sb = (size_t)(b * NCH + c) * 128 + d;
    Sin[sb * 128 + e] = s;
    s = fmaf(Pc[sb], s, Sc[sb * 128 + e]);
  }
}

// ---------------------------------------------------------------------------
// Pass 3: correction + residual.  T_{-1} = Sin, T_t = g_t (*) T_{t-1},
//   xout = y_local + q_t . T_t + xin
// Same block mapping as pass 1.
// ---------------------------------------------------------------------------
__global__ void k_scan3(const float* __restrict__ qkv,
                        const float* __restrict__ g,
                        const float* __restrict__ Sin,
                        const float* __restrict__ ybuf,
                        const float* __restrict__ xin,
                        float* __restrict__ xout) {
  const int eg = blockIdx.x & 15;
  const int c  = (blockIdx.x >> 4) & (NCH - 1);
  const int b  = blockIdx.x >> 9;
  const int wv = threadIdx.x >> 5;
  const int ln = threadIdx.x & 31;
  const int e  = eg * 8 + wv;
  const int d0 = ln * 4;
  const size_t sb = (size_t)(b * NCH + c) * 128;
  float t0 = Sin[(sb + d0 + 0) * 128 + e];
  float t1 = Sin[(sb + d0 + 1) * 128 + e];
  float t2 = Sin[(sb + d0 + 2) * 128 + e];
  float t3 = Sin[(sb + d0 + 3) * 128 + e];
  const size_t rbase = (size_t)b * SEQ + (size_t)c * CHK;
  for (int t = 0; t < CHK; ++t) {
    const float* r  = qkv + (rbase + t) * 384;
    const float* gr = g   + (rbase + t) * 128;
    float4 qv = *(const float4*)(r + d0);
    float4 gv = *(const float4*)(gr + d0);
    t0 *= gv.x; t1 *= gv.y; t2 *= gv.z; t3 *= gv.w;
    float acc = qv.x * t0 + qv.y * t1 + qv.z * t2 + qv.w * t3;
    acc += __shfl_xor(acc, 1, 32);
    acc += __shfl_xor(acc, 2, 32);
    acc += __shfl_xor(acc, 4, 32);
    acc += __shfl_xor(acc, 8, 32);
    acc += __shfl_xor(acc, 16, 32);
    if (ln == 0) {
      size_t oi = (rbase + t) * 128 + e;
      xout[oi] = acc + ybuf[oi] + xin[oi];
    }
  }
}

// ---------------------------------------------------------------------------
// Host-side orchestration
// ---------------------------------------------------------------------------
extern "C" void kernel_launch(void* const* d_in, const int* in_sizes, int n_in,
                              void* d_out, int out_size, void* d_ws, size_t ws_size,
                              hipStream_t stream) {
  (void)in_sizes; (void)n_in; (void)out_size; (void)ws_size;

  const int*   tokens     = (const int*)  d_in[0];
  const float* emb        = (const float*)d_in[1];
  const float* gamma_attn = (const float*)d_in[2];
  const float* Wqkv       = (const float*)d_in[3];
  const float* Wa         = (const float*)d_in[4];
  const float* ba         = (const float*)d_in[5];
  const float* gamma_ff   = (const float*)d_in[6];
  const float* W1         = (const float*)d_in[7];
  const float* b1         = (const float*)d_in[8];
  const float* W2         = (const float*)d_in[9];
  const float* b2         = (const float*)d_in[10];
  const float* gamma_out  = (const float*)d_in[11];
  const float* Wlogits    = (const float*)d_in[12];
  float* out = (float*)d_out;

  // ---- workspace carve-up (256B aligned) ----
  char* base = (char*)d_ws;
  size_t off = 0;
  auto carve = [&](size_t bytes) -> char* {
    char* p = base + off;
    off = (off + bytes + 255) & ~(size_t)255;
    return p;
  };
  float*    xA    = (float*)   carve((size_t)MROWS * DIM * 4);
  float*    xB    = (float*)   carve((size_t)MROWS * DIM * 4);
  _Float16* h16   = (_Float16*)carve((size_t)MROWS * DIM * 2);
  float*    qkv   = (float*)   carve((size_t)MROWS * 384 * 4);
  float*    abuf  = (float*)   carve((size_t)MROWS * 256 * 4);
  float*    gbuf  = (float*)   carve((size_t)MROWS * DIM * 4);
  _Float16* ff1h  = (_Float16*)carve((size_t)MROWS * FFD * 2);
  float*    ybuf  = (float*)   carve((size_t)MROWS * DIM * 4);
  float*    Sc    = (float*)   carve((size_t)BATCH * NCH * DIM * DIM * 4);
  float*    Sin   = (float*)   carve((size_t)BATCH * NCH * DIM * DIM * 4);
  float*    Pc    = (float*)   carve((size_t)BATCH * NCH * DIM * 4);
  _Float16* pQKV[DEPTH], *pA[DEPTH], *pW1[DEPTH], *pW2[DEPTH];
  for (int l = 0; l < DEPTH; ++l) {
    pQKV[l] = (_Float16*)carve((size_t)DIM * 3 * DIM * 2);
    pA[l]   = (_Float16*)carve((size_t)DIM * 2 * DIM * 2);
    pW1[l]  = (_Float16*)carve((size_t)DIM * FFD * 2);
    pW2[l]  = (_Float16*)carve((size_t)FFD * DIM * 2);
  }
  _Float16* pL = (_Float16*)carve((size_t)DIM * VOCAB * 2);

  auto packLaunch = [&](const float* W, _Float16* dst, int K, int N) {
    int total = (N >> 4) * (K >> 5) * 256;           // dwords
    k_packB<<<(total + 255) / 256, 256, 0, stream>>>(W, dst, K, N);
  };

  // ---- pack all weights to f16 WMMA-B layout ----
  for (int l = 0; l < DEPTH; ++l) {
    packLaunch(Wqkv + (size_t)l * DIM * 3 * DIM, pQKV[l], DIM, 3 * DIM);
    packLaunch(Wa   + (size_t)l * DIM * 2 * DIM, pA[l],   DIM, 2 * DIM);
    packLaunch(W1   + (size_t)l * DIM * FFD,     pW1[l],  DIM, FFD);
    packLaunch(W2   + (size_t)l * FFD * DIM,     pW2[l],  FFD, DIM);
  }
  packLaunch(Wlogits, pL, DIM, VOCAB);

  // ---- embedding ----
  k_embed<<<(MROWS * (DIM / 4) + 255) / 256, 256, 0, stream>>>(tokens, emb, xA);

  const dim3 blk(128);
  float* xcur = xA;
  float* xnxt = xB;

  for (int l = 0; l < DEPTH; ++l) {
    // ----- attention branch -----
    k_rmsnorm<<<MROWS / 8, 256, 0, stream>>>(xcur, gamma_attn + (size_t)l * DIM, h16);
    k_gemm<1><<<dim3((3 * DIM) / 64, MROWS / 16), blk, 0, stream>>>(
        h16, pQKV[l], nullptr, nullptr, qkv, nullptr, MROWS, 3 * DIM, DIM, EPI_F32);
    k_gemm<4><<<dim3((2 * DIM) / 256, MROWS / 16), blk, 0, stream>>>(
        h16, pA[l], ba + (size_t)l * 2 * DIM, nullptr, abuf, nullptr,
        MROWS, 2 * DIM, DIM, EPI_F32 | EPI_BIAS);
    k_gate<<<(MROWS * DIM + 255) / 256, 256, 0, stream>>>(abuf, gbuf);
    // chunked scan: local pass, chunk combine, correction (+residual)
    k_scan1<<<BATCH * NCH * 16, 256, 0, stream>>>(qkv, gbuf, ybuf, Sc, Pc);
    k_scan2<<<(BATCH * DIM * DIM + 255) / 256, 256, 0, stream>>>(Sc, Pc, Sin);
    k_scan3<<<BATCH * NCH * 16, 256, 0, stream>>>(qkv, gbuf, Sin, ybuf, xcur, xnxt);
    { float* t = xcur; xcur = xnxt; xnxt = t; }

    // ----- feed-forward branch -----
    k_rmsnorm<<<MROWS / 8, 256, 0, stream>>>(xcur, gamma_ff + (size_t)l * DIM, h16);
    k_gemm<4><<<dim3(FFD / 256, MROWS / 16), blk, 0, stream>>>(
        h16, pW1[l], b1 + (size_t)l * FFD, nullptr, nullptr, ff1h,
        MROWS, FFD, DIM, EPI_BIAS | EPI_GELU | EPI_F16);
    k_gemm<1><<<dim3(DIM / 64, MROWS / 16), blk, 0, stream>>>(
        ff1h, pW2[l], b2 + (size_t)l * DIM, xcur, xnxt, nullptr,
        MROWS, DIM, FFD, EPI_BIAS | EPI_RES | EPI_F32);
    { float* t = xcur; xcur = xnxt; xnxt = t; }
  }

  // ---- logits ----
  k_rmsnorm<<<MROWS / 8, 256, 0, stream>>>(xcur, gamma_out, h16);
  k_gemm<4><<<dim3(VOCAB / 256, MROWS / 16), blk, 0, stream>>>(
      h16, pL, nullptr, nullptr, out, nullptr, MROWS, VOCAB, DIM, EPI_F32);
}
